// MultiHeadsAttention_62929860821177
// MI455X (gfx1250) — matrix operand order repaired and checked by reading, not compiled, not run on hardware
//
#include <hip/hip_runtime.h>

// ---------------------------------------------------------------------------
// MI455X (gfx1250) fused multi-head attention, f16 WMMA pipeline.
//   1) Q/K/V projections : fp32 x/W -> f16 WMMA GEMM -> f16 workspace
//   2) flash-attention   : fused QK^T + rel_pos + causal softmax + PV
//                          with double-buffered ASYNC global->LDS staging
//   3) output projection : f16 O @ Wo + bo -> fp32 out
// ---------------------------------------------------------------------------

#define EMBED   1024
#define HEADS   16
#define HEAD_DIM 64
#define BATCH   2
#define SEQ     2048
#define MROWS   (BATCH * SEQ)        // 4096

typedef __attribute__((ext_vector_type(16))) _Float16 v16h;
typedef __attribute__((ext_vector_type(8)))  float    v8f;
typedef __attribute__((ext_vector_type(4)))  unsigned int u32x4;
typedef __attribute__((ext_vector_type(4)))  int          v4i;

union Frag {            // 16 f16 = 8 VGPRs, matches WMMA A/B operand
    v16h  h;
    u32x4 q[2];
};

__device__ __forceinline__ v8f wmma_f16(const v16h& a, const v16h& b, const v8f& c) {
    return __builtin_amdgcn_wmma_f32_16x16x32_f16(
        /*neg_a=*/false, a, /*neg_b=*/false, b,
        /*c_mod=*/(short)0, c, /*reuse_a=*/false, /*reuse_b=*/false);
}

// ---- async global->LDS (ASYNCcnt-tracked) --------------------------------
#define AS1 __attribute__((address_space(1)))
#define AS3 __attribute__((address_space(3)))

__device__ __forceinline__ void async_g2l_b128(const void* g, void* l) {
#if __has_builtin(__builtin_amdgcn_global_load_async_to_lds_b128)
    __builtin_amdgcn_global_load_async_to_lds_b128(
        (AS1 v4i*)(void*)g, (AS3 v4i*)l, 0, 0);
#else
    unsigned loff = (unsigned)(unsigned long long)l;
    asm volatile("global_load_async_to_lds_b128 %0, %1, off"
                 :: "v"(loff), "v"(g) : "memory");
#endif
}

#if __has_builtin(__builtin_amdgcn_s_wait_asynccnt)
#define WAIT_ASYNC(n) __builtin_amdgcn_s_wait_asynccnt(n)
#else
#define WAIT_ASYNC(n) asm volatile("s_wait_asynccnt %0" :: "i"(n) : "memory")
#endif

// ---------------------------------------------------------------------------
// GEMM: C[M x N] = A[M x K] @ W[K x N] + bias     (tiles 128x128x32, 8 waves)
// ---------------------------------------------------------------------------
template <typename TA, typename TC>
__global__ __launch_bounds__(256)
void gemm_bias_wmma(const TA* __restrict__ A, const float* __restrict__ W,
                    const float* __restrict__ bias, TC* __restrict__ C,
                    int M, int N, int K)
{
    __shared__ _Float16 As[128 * 48];   // [m][k], stride 96 B
    __shared__ _Float16 Bs[128 * 48];   // [n][k], stride 96 B

    const int tid   = threadIdx.x;
    const int lane  = tid & 31;
    const int wave  = tid >> 5;
    const int wm    = wave >> 1;
    const int wn    = wave & 1;
    const int lrow  = lane & 15;
    const int lhalf = lane >> 4;
    const int m0    = blockIdx.y * 128;
    const int n0    = blockIdx.x * 128;

    v8f acc[2][4];
#pragma unroll
    for (int mt = 0; mt < 2; ++mt)
#pragma unroll
        for (int nt = 0; nt < 4; ++nt) acc[mt][nt] = {};

    for (int k0 = 0; k0 < K; k0 += 32) {
        for (int idx = tid; idx < 128 * 32; idx += 256) {
            int m = idx >> 5, kk = idx & 31;
            As[m * 48 + kk] = (_Float16)(float)A[(size_t)(m0 + m) * K + k0 + kk];
        }
        for (int idx = tid; idx < 32 * 128; idx += 256) {
            int kk = idx >> 7, n = idx & 127;
            Bs[n * 48 + kk] = (_Float16)W[(size_t)(k0 + kk) * N + n0 + n];
        }
        __syncthreads();

        Frag a[2];
#pragma unroll
        for (int mt = 0; mt < 2; ++mt) {
            const _Float16* ar = &As[(wm * 32 + mt * 16 + lrow) * 48];
            a[mt].q[0] = *(const u32x4*)(ar + lhalf * 8);
            a[mt].q[1] = *(const u32x4*)(ar + 16 + lhalf * 8);
        }
#pragma unroll
        for (int nt = 0; nt < 4; ++nt) {
            Frag b;
            const _Float16* br = &Bs[(wn * 64 + nt * 16 + lrow) * 48];
            b.q[0] = *(const u32x4*)(br + lhalf * 8);
            b.q[1] = *(const u32x4*)(br + 16 + lhalf * 8);
#pragma unroll
            for (int mt = 0; mt < 2; ++mt)
                acc[mt][nt] = wmma_f16(a[mt].h, b.h, acc[mt][nt]);
        }
        __syncthreads();
    }

#pragma unroll
    for (int mt = 0; mt < 2; ++mt)
#pragma unroll
        for (int nt = 0; nt < 4; ++nt)
#pragma unroll
            for (int r = 0; r < 8; ++r) {
                int row = m0 + wm * 32 + mt * 16 + (lhalf ? r + 8 : r);
                int col = n0 + wn * 64 + nt * 16 + lrow;
                float v = acc[mt][nt][r] + bias[col];
                C[(size_t)row * N + col] = (TC)v;
            }
}

// ---------------------------------------------------------------------------
// Fused causal attention with rel_pos bias, double-buffered async staging.
// Grid: (SEQ/64, BATCH*HEADS).  Block: 128 threads = 4 waves, 16 q-rows/wave.
// ---------------------------------------------------------------------------
__global__ __launch_bounds__(128)
void attention_wmma(const _Float16* __restrict__ Qh,
                    const _Float16* __restrict__ Kh,
                    const _Float16* __restrict__ Vh,
                    const float*    __restrict__ rel_pos,
                    _Float16*       __restrict__ Oh)
{
    __shared__ _Float16 Ks[2][64][72];     // [j][d]   (async-staged, raw copy)
    __shared__ _Float16 Vs[2][64][72];     // [d][j]   (sync transpose)
    __shared__ float    RPs[2][64][68];    // [i][j]   (async-staged fp32)
    __shared__ _Float16 Ps[4][16][72];     // per-wave P tile [row][j]

    const int tid   = threadIdx.x;
    const int lane  = tid & 31;
    const int w     = tid >> 5;
    const int lrow  = lane & 15;
    const int lhalf = lane >> 4;

    const int i0 = blockIdx.x * 64;
    const int bh = blockIdx.y;
    const int b  = bh / HEADS;
    const int h  = bh % HEADS;

    const float NEG_INF = -1.0e30f;
    const float scaling = 0.125f;          // HEAD_DIM^-0.5

    const _Float16* kbase = Kh + (size_t)b * SEQ * EMBED + h * HEAD_DIM;
    const _Float16* vbase = Vh + (size_t)b * SEQ * EMBED + h * HEAD_DIM;
    const float*  rp_base = rel_pos + ((size_t)(b * HEADS + h)) * SEQ * SEQ;

    // ---- this wave's Q fragments (16 rows x 64 d), kept in registers
    Frag qf[2];
    {
        const _Float16* qrow =
            Qh + ((size_t)(b * SEQ + i0 + w * 16 + lrow)) * EMBED + h * HEAD_DIM;
        qf[0].q[0] = *(const u32x4*)(qrow + lhalf * 8);
        qf[0].q[1] = *(const u32x4*)(qrow + 16 + lhalf * 8);
        qf[1].q[0] = *(const u32x4*)(qrow + 32 + lhalf * 8);
        qf[1].q[1] = *(const u32x4*)(qrow + 48 + lhalf * 8);
    }

    v8f   acc_o[4];
    float m_run[8], l_run[8];
#pragma unroll
    for (int dt = 0; dt < 4; ++dt) acc_o[dt] = {};
#pragma unroll
    for (int r = 0; r < 8; ++r) { m_run[r] = NEG_INF; l_run[r] = 0.0f; }

    // ---- staging helpers: 12 async b128 ops / thread / chunk --------------
    auto issue_async = [&](int j0, int buf) {
#pragma unroll
        for (int qq = 0; qq < 4; ++qq) {            // K tile: 64x64 f16 = 8 KB
            int idx = qq * 128 + tid;
            int jj = idx >> 3, c = idx & 7;
            async_g2l_b128(kbase + (size_t)(j0 + jj) * EMBED + c * 8,
                           &Ks[buf][jj][c * 8]);
        }
#pragma unroll
        for (int qq = 0; qq < 8; ++qq) {            // rel_pos tile: 64x64 f32 = 16 KB
            int idx = qq * 128 + tid;
            int row = idx >> 4, c = idx & 15;
            async_g2l_b128(rp_base + (size_t)(i0 + row) * SEQ + j0 + c * 4,
                           &RPs[buf][row][c * 4]);
        }
    };
    auto stage_v = [&](int j0, int buf) {           // V transpose: Vs[d][j]
        for (int idx = tid; idx < 64 * 64; idx += 128) {
            int dd = idx & 63, jj = idx >> 6;
            Vs[buf][dd][jj] = vbase[(size_t)(j0 + jj) * EMBED + dd];
        }
    };

    const int nch = i0 / 64 + 1;                    // causal chunk count
    issue_async(0, 0);
    stage_v(0, 0);

    for (int c = 0; c < nch; ++c) {
        const int cur = c & 1;
        const int j0  = c * 64;
        const bool more = (c + 1) < nch;
        if (more) {                                  // prefetch next chunk
            issue_async(j0 + 64, cur ^ 1);
            stage_v(j0 + 64, cur ^ 1);
            WAIT_ASYNC(12);                          // drain current chunk only
        } else {
            WAIT_ASYNC(0);
        }
        __syncthreads();

        // ---- S = Q @ K^T
        v8f s[4];
#pragma unroll
        for (int jt = 0; jt < 4; ++jt) {
            Frag k0f, k1f;
            const _Float16* krow = &Ks[cur][jt * 16 + lrow][0];
            k0f.q[0] = *(const u32x4*)(krow + lhalf * 8);
            k0f.q[1] = *(const u32x4*)(krow + 16 + lhalf * 8);
            k1f.q[0] = *(const u32x4*)(krow + 32 + lhalf * 8);
            k1f.q[1] = *(const u32x4*)(krow + 48 + lhalf * 8);
            v8f cacc = {};
            cacc = wmma_f16(qf[0].h, k0f.h, cacc);
            cacc = wmma_f16(qf[1].h, k1f.h, cacc);
            s[jt] = cacc;
        }

        // ---- + rel_pos (from LDS), scale, causal mask
#pragma unroll
        for (int jt = 0; jt < 4; ++jt)
#pragma unroll
            for (int r = 0; r < 8; ++r) {
                int rl   = w * 16 + (lhalf ? r + 8 : r);
                int irow = i0 + rl;
                int jcol = j0 + jt * 16 + lrow;
                float v = (s[jt][r] + RPs[cur][rl][jt * 16 + lrow]) * scaling;
                if (jcol > irow) v = NEG_INF;
                s[jt][r] = v;
            }

        // ---- online softmax (rows across 16 lanes of one half-wave)
#pragma unroll
        for (int r = 0; r < 8; ++r) {
            float tmax = s[0][r];
#pragma unroll
            for (int jt = 1; jt < 4; ++jt) tmax = fmaxf(tmax, s[jt][r]);
#pragma unroll
            for (int off = 8; off >= 1; off >>= 1)
                tmax = fmaxf(tmax, __shfl_xor(tmax, off, 32));
            float m_new = fmaxf(m_run[r], tmax);
            float resc  = __expf(m_run[r] - m_new);
            float rsum  = 0.0f;
#pragma unroll
            for (int jt = 0; jt < 4; ++jt) {
                float p = __expf(s[jt][r] - m_new);
                s[jt][r] = p;
                rsum += p;
            }
#pragma unroll
            for (int off = 8; off >= 1; off >>= 1)
                rsum += __shfl_xor(rsum, off, 32);
            l_run[r] = l_run[r] * resc + rsum;
            m_run[r] = m_new;
#pragma unroll
            for (int dt = 0; dt < 4; ++dt) acc_o[dt][r] *= resc;
        }

        // ---- P -> LDS (A-frag layout)
#pragma unroll
        for (int jt = 0; jt < 4; ++jt)
#pragma unroll
            for (int r = 0; r < 8; ++r)
                Ps[w][lhalf ? r + 8 : r][jt * 16 + lrow] = (_Float16)s[jt][r];

        Frag pa0, pa1;
        {
            const _Float16* prow = &Ps[w][lrow][0];
            pa0.q[0] = *(const u32x4*)(prow + lhalf * 8);
            pa0.q[1] = *(const u32x4*)(prow + 16 + lhalf * 8);
            pa1.q[0] = *(const u32x4*)(prow + 32 + lhalf * 8);
            pa1.q[1] = *(const u32x4*)(prow + 48 + lhalf * 8);
        }

        // ---- O += P @ V
#pragma unroll
        for (int dt = 0; dt < 4; ++dt) {
            Frag v0f, v1f;
            const _Float16* vrow = &Vs[cur][dt * 16 + lrow][0];
            v0f.q[0] = *(const u32x4*)(vrow + lhalf * 8);
            v0f.q[1] = *(const u32x4*)(vrow + 16 + lhalf * 8);
            v1f.q[0] = *(const u32x4*)(vrow + 32 + lhalf * 8);
            v1f.q[1] = *(const u32x4*)(vrow + 48 + lhalf * 8);
            acc_o[dt] = wmma_f16(pa0.h, v0f.h, acc_o[dt]);
            acc_o[dt] = wmma_f16(pa1.h, v1f.h, acc_o[dt]);
        }
        __syncthreads();            // all reads of [cur] done before next issue
    }

    // ---- normalize and write O (f16, [b*S + i][h*64 + d])
#pragma unroll
    for (int dt = 0; dt < 4; ++dt)
#pragma unroll
        for (int r = 0; r < 8; ++r) {
            int irow = i0 + w * 16 + (lhalf ? r + 8 : r);
            int dd   = dt * 16 + lrow;
            float v = acc_o[dt][r] / l_run[r];
            Oh[((size_t)(b * SEQ + irow)) * EMBED + h * HEAD_DIM + dd] = (_Float16)v;
        }
}

// ---------------------------------------------------------------------------
// Launch
// ---------------------------------------------------------------------------
extern "C" void kernel_launch(void* const* d_in, const int* in_sizes, int n_in,
                              void* d_out, int out_size, void* d_ws, size_t ws_size,
                              hipStream_t stream)
{
    const float* x       = (const float*)d_in[0];
    const float* rel_pos = (const float*)d_in[1];
    const float* Wq      = (const float*)d_in[2];
    const float* bq      = (const float*)d_in[3];
    const float* Wk      = (const float*)d_in[4];
    const float* bk      = (const float*)d_in[5];
    const float* Wv      = (const float*)d_in[6];
    const float* bv      = (const float*)d_in[7];
    const float* Wo      = (const float*)d_in[8];
    const float* bo      = (const float*)d_in[9];
    float* out = (float*)d_out;

    _Float16* Qh = (_Float16*)d_ws;
    _Float16* Kh = Qh + (size_t)MROWS * EMBED;
    _Float16* Vh = Kh + (size_t)MROWS * EMBED;
    _Float16* Oh = Vh + (size_t)MROWS * EMBED;

    dim3 ggrid(EMBED / 128, MROWS / 128);   // (8, 32)
    dim3 gblk(256);

    gemm_bias_wmma<float, _Float16><<<ggrid, gblk, 0, stream>>>(x, Wq, bq, Qh, MROWS, EMBED, EMBED);
    gemm_bias_wmma<float, _Float16><<<ggrid, gblk, 0, stream>>>(x, Wk, bk, Kh, MROWS, EMBED, EMBED);
    gemm_bias_wmma<float, _Float16><<<ggrid, gblk, 0, stream>>>(x, Wv, bv, Vh, MROWS, EMBED, EMBED);

    dim3 agrid(SEQ / 64, BATCH * HEADS);    // (32, 32)
    dim3 ablk(128);
    attention_wmma<<<agrid, ablk, 0, stream>>>(Qh, Kh, Vh, rel_pos, Oh);

    gemm_bias_wmma<_Float16, float><<<ggrid, gblk, 0, stream>>>(Oh, Wo, bo, out, MROWS, EMBED, EMBED);
}
